// ImageSelfAttention_28759101014192
// MI455X (gfx1250) — matrix operand order repaired and checked by reading, not compiled, not run on hardware
//
#include <hip/hip_runtime.h>

// ---------------------------------------------------------------------------
// ImageSelfAttention for MI455X (gfx1250, wave32, WMMA).
//   B=32, C=512, N=H*W=1024, INNER=64.  All GEMMs: v_wmma_f32_16x16x32_f16.
// Every wave computes a 16x64 / 64x16 macro-tile: 4 WMMA accumulators sharing
// one operand -> 4 WMMAs per 10 b128 loads per k-step, zero VALU in K-loops.
// Workspace layout (needs ~136.7 MiB):
//   [  0, 32MiB)  Xt  = X^T  f16 [B][N][C]   (contiguous B operand for K1)
//   [ 32, 36MiB)  Ft  = F^T  f16 [B][N][64]
//   [ 36, 40MiB)  Gt  = G^T  f16 [B][N][64]
//   [ 40, 72MiB)  Hm  = H    f16 [B][C][N]
//   [ 72,136MiB)  St  = sim^T f16 [B][N(m)][N(n)], softmax'd in place -> P^T
//   [136MiB, +640KiB) Wc = [Wf;Wg;Wh] f16 [640][512] (pre-converted weights)
// ---------------------------------------------------------------------------

typedef _Float16 v8h  __attribute__((ext_vector_type(8)));
typedef _Float16 v16h __attribute__((ext_vector_type(16)));
typedef float    v8f  __attribute__((ext_vector_type(8)));

#define BATCH 32
#define CDIM  512
#define NDIM  1024
#define IDIM  64
#define ROWS  (IDIM + IDIM + CDIM)   // 640 concat rows of [Wf;Wg;Wh]

static __device__ __forceinline__ v8f wmma16x16x32(v16h a, v16h b, v8f c) {
  // (neg_a, A, neg_b, B, c_mod, C, reuse_a, reuse_b)
  return __builtin_amdgcn_wmma_f32_16x16x32_f16(false, a, false, b,
                                                (short)0, c, false, false);
}

// A-operand lane layout (ISA 7.12.2, 16-bit A 16x32):
//   lane<16 : K = {k0..k0+7, k0+16..k0+23}; lane>=16: shifted by +8.
static __device__ __forceinline__ v16h load_a_f16(const _Float16* row, int k0, int hl) {
  const int ka = k0 + 8 * hl;
  v8h lo = *(const v8h*)(row + ka);
  v8h hi = *(const v8h*)(row + ka + 16);
  v16h a;
#pragma unroll
  for (int j = 0; j < 8; ++j) { a[j] = lo[j]; a[8 + j] = hi[j]; }
  return a;
}

// B-operand lane layout (same K order as A, 32x16):
//   lane<16 : K = {k0..k0+7, k0+16..k0+23}; lane>=16: shifted by +8.
static __device__ __forceinline__ v16h load_b_f16(const _Float16* col, int k0, int hl) {
  const int kb = k0 + 8 * hl;
  v8h lo = *(const v8h*)(col + kb);
  v8h hi = *(const v8h*)(col + kb + 16);
  v16h b;
#pragma unroll
  for (int j = 0; j < 8; ++j) { b[j] = lo[j]; b[8 + j] = hi[j]; }
  return b;
}

static __device__ __forceinline__ v8h pack_f16(v8f acc) {
  v8h p;
#pragma unroll
  for (int r = 0; r < 8; ++r) p[r] = (_Float16)acc[r];
  return p;
}

// ---------------------------------------------------------------------------
// K0a: coalesced transpose+convert  x fp32 [B][C][N] -> Xt f16 [B][N][C].
// 32x32 tiles through LDS (stride 34 halves = 17 banks: conflict-free).
// ---------------------------------------------------------------------------
__global__ void k0_transpose(const float* __restrict__ x,
                             _Float16* __restrict__ Xt) {
  __shared__ _Float16 lds[32][34];
  const int tx = threadIdx.x & 31;        // fast axis
  const int ty = threadIdx.x >> 5;        // 0..7

  const int tiles_per_b = (CDIM / 32) * (NDIM / 32);   // 16*32
  const int b  = blockIdx.x / tiles_per_b;
  const int r2 = blockIdx.x % tiles_per_b;
  const int c0 = (r2 / (NDIM / 32)) * 32;
  const int n0 = (r2 % (NDIM / 32)) * 32;

#pragma unroll
  for (int r = 0; r < 4; ++r) {           // read: contiguous over n
    const int c = c0 + ty + 8 * r;
    lds[ty + 8 * r][tx] = (_Float16)x[((size_t)b * CDIM + c) * NDIM + n0 + tx];
  }
  __syncthreads();
#pragma unroll
  for (int r = 0; r < 4; ++r) {           // write: contiguous over c
    const int n = n0 + ty + 8 * r;
    Xt[((size_t)b * NDIM + n) * CDIM + c0 + tx] = lds[tx][ty + 8 * r];
  }
}

// ---------------------------------------------------------------------------
// K0b: one-shot weight conversion [Wf;Wg;Wh] fp32 -> Wc f16 [640][512].
// Done once so K1's inner loop never touches fp32 / v_cvt again.
// ---------------------------------------------------------------------------
__global__ void k0_wconv(const float* __restrict__ Wf,
                         const float* __restrict__ Wg,
                         const float* __restrict__ Wh,
                         _Float16* __restrict__ Wc) {
  const int e = blockIdx.x * 256 + threadIdx.x;        // 0 .. 640*512-1
  float v;
  if (e < IDIM * CDIM)               v = Wf[e];
  else if (e < 2 * IDIM * CDIM)      v = Wg[e - IDIM * CDIM];
  else                               v = Wh[e - 2 * IDIM * CDIM];
  Wc[e] = (_Float16)v;
}

// ---------------------------------------------------------------------------
// K1: projections.  Each wave: 64 rows of Wc x 16 cols of X (4 WMMA accs
// sharing the X operand).  Row-block rb==0 -> F, rb==1 -> G, rb>=2 -> H.
// ---------------------------------------------------------------------------
__global__ void k1_proj(const _Float16* __restrict__ Wc,
                        const _Float16* __restrict__ Xt,
                        _Float16* __restrict__ Ft,
                        _Float16* __restrict__ Gt,
                        _Float16* __restrict__ Hm) {
  const int wave = threadIdx.x >> 5;
  const int lane = threadIdx.x & 31;
  const int hl   = lane >> 4;
  const int l16  = lane & 15;

  const int tiles_per_b = (ROWS / 64) * (NDIM / 16);   // 10*64
  int t  = blockIdx.x * 4 + wave;
  int b  = t / tiles_per_b;
  int r2 = t % tiles_per_b;
  int rb = r2 / (NDIM / 16);              // 64-row block: 0=F 1=G 2..9=H
  int nt = r2 % (NDIM / 16);

  const int nn = nt * 16 + l16;
  const _Float16* ar[4];
#pragma unroll
  for (int s = 0; s < 4; ++s)
    ar[s] = Wc + (size_t)(rb * 64 + s * 16 + l16) * CDIM;
  const _Float16* brow = Xt + ((size_t)b * NDIM + nn) * CDIM;

  v8f acc[4] = {{}, {}, {}, {}};
  for (int k0 = 0; k0 < CDIM; k0 += 32) {
    __builtin_prefetch(brow + k0 + 128, 0, 0);         // global_prefetch_b8
    const v16h bm = load_b_f16(brow, k0, hl);          // shared across 4 tiles
#pragma unroll
    for (int s = 0; s < 4; ++s)
      acc[s] = wmma16x16x32(load_a_f16(ar[s], k0, hl), bm, acc[s]);
  }

  if (rb == 0) {                          // F -> Ft[b][n][i], i contiguous
#pragma unroll
    for (int s = 0; s < 4; ++s)
      *(v8h*)(Ft + ((size_t)b * NDIM + nn) * IDIM + s * 16 + 8 * hl) = pack_f16(acc[s]);
  } else if (rb == 1) {                   // G -> Gt[b][m][i]
#pragma unroll
    for (int s = 0; s < 4; ++s)
      *(v8h*)(Gt + ((size_t)b * NDIM + nn) * IDIM + s * 16 + 8 * hl) = pack_f16(acc[s]);
  } else {                                // H -> Hm[b][c][n]
#pragma unroll
    for (int s = 0; s < 4; ++s) {
      const int c0 = (rb - 2) * 64 + s * 16 + 8 * hl;
#pragma unroll
      for (int r = 0; r < 8; ++r)
        Hm[((size_t)b * CDIM + c0 + r) * NDIM + nn] = (_Float16)acc[s][r];
    }
  }
}

// ---------------------------------------------------------------------------
// K2: sim[n][m] = sum_i Ft[n][i]*Gt[m][i]  (K=64 -> 2 wmma steps).
// Wave: 16 n x 64 m, A operand shared.  Stored TRANSPOSED as St[b][m][n]
// (packed 16B stores; softmax axis becomes row-contiguous).
// ---------------------------------------------------------------------------
__global__ void k2_sim(const _Float16* __restrict__ Ft,
                       const _Float16* __restrict__ Gt,
                       _Float16* __restrict__ St) {
  const int wave = threadIdx.x >> 5;
  const int lane = threadIdx.x & 31;
  const int hl   = lane >> 4;
  const int l16  = lane & 15;

  const int tiles_per_b = (NDIM / 16) * (NDIM / 64);   // 64*16
  int t  = blockIdx.x * 4 + wave;
  int b  = t / tiles_per_b;
  int r2 = t % tiles_per_b;
  int nt = r2 / (NDIM / 64);
  int mb = r2 % (NDIM / 64);

  const _Float16* ar = Ft + ((size_t)b * NDIM + nt * 16 + l16) * IDIM;
  const _Float16* br[4];
#pragma unroll
  for (int s = 0; s < 4; ++s)
    br[s] = Gt + ((size_t)b * NDIM + mb * 64 + s * 16 + l16) * IDIM;

  v8f acc[4] = {{}, {}, {}, {}};
#pragma unroll
  for (int k0 = 0; k0 < IDIM; k0 += 32) {
    const v16h a = load_a_f16(ar, k0, hl);             // shared across 4 tiles
#pragma unroll
    for (int s = 0; s < 4; ++s)
      acc[s] = wmma16x16x32(a, load_b_f16(br[s], k0, hl), acc[s]);
  }

#pragma unroll
  for (int s = 0; s < 4; ++s) {
    const int m = mb * 64 + s * 16 + l16;
    *(v8h*)(St + ((size_t)b * NDIM + m) * NDIM + nt * 16 + 8 * hl) = pack_f16(acc[s]);
  }
}

// ---------------------------------------------------------------------------
// K3: in-place softmax over each St row (the n axis == reference axis=1).
// One wave32 per row; cross-lane max/sum via __shfl_xor (wave32).
// ---------------------------------------------------------------------------
__global__ void k3_softmax(_Float16* __restrict__ St) {
  const int wave = threadIdx.x >> 5;
  const int lane = threadIdx.x & 31;
  _Float16* row = St + (size_t)(blockIdx.x * 8 + wave) * NDIM;

  float v[32];
  float mx = -3.4e38f;
#pragma unroll
  for (int j = 0; j < 32; ++j) {
    v[j] = (float)row[j * 32 + lane];
    mx = fmaxf(mx, v[j]);
  }
#pragma unroll
  for (int off = 16; off; off >>= 1) mx = fmaxf(mx, __shfl_xor(mx, off, 32));
  float s = 0.f;
#pragma unroll
  for (int j = 0; j < 32; ++j) { v[j] = __expf(v[j] - mx); s += v[j]; }
#pragma unroll
  for (int off = 16; off; off >>= 1) s += __shfl_xor(s, off, 32);
  const float inv = __frcp_rn(s);
#pragma unroll
  for (int j = 0; j < 32; ++j) row[j * 32 + lane] = (_Float16)(v[j] * inv);
}

// ---------------------------------------------------------------------------
// K4: O[c][m] = sum_n Hm[c][n] * P[n][m], P[n][m] = St[m][n] (post-softmax).
// Wave: 16 c x 64 m, A operand (H row) shared.  K=1024, all b128 loads.
// ---------------------------------------------------------------------------
__global__ void k4_out(const _Float16* __restrict__ Hm,
                       const _Float16* __restrict__ P,
                       float* __restrict__ out) {
  const int wave = threadIdx.x >> 5;
  const int lane = threadIdx.x & 31;
  const int hl   = lane >> 4;
  const int l16  = lane & 15;

  const int tiles_per_b = (CDIM / 16) * (NDIM / 64);   // 32*16
  int t  = blockIdx.x * 4 + wave;
  int b  = t / tiles_per_b;
  int r2 = t % tiles_per_b;
  int ct = r2 / (NDIM / 64);
  int mb = r2 % (NDIM / 64);

  const _Float16* ar = Hm + ((size_t)b * CDIM + ct * 16 + l16) * NDIM;
  const _Float16* br[4];
#pragma unroll
  for (int s = 0; s < 4; ++s)
    br[s] = P + ((size_t)b * NDIM + mb * 64 + s * 16 + l16) * NDIM;

  v8f acc[4] = {{}, {}, {}, {}};
  for (int k0 = 0; k0 < NDIM; k0 += 32) {
    __builtin_prefetch(ar + k0 + 256, 0, 0);
    const v16h a = load_a_f16(ar, k0, hl);             // shared across 4 tiles
#pragma unroll
    for (int s = 0; s < 4; ++s)
      acc[s] = wmma16x16x32(a, load_b_f16(br[s], k0, hl), acc[s]);
  }

  const int c0 = ct * 16 + 8 * hl;
#pragma unroll
  for (int s = 0; s < 4; ++s)
#pragma unroll
    for (int r = 0; r < 8; ++r)
      out[((size_t)b * CDIM + c0 + r) * NDIM + mb * 64 + s * 16 + l16] = acc[s][r];
}

// ---------------------------------------------------------------------------
extern "C" void kernel_launch(void* const* d_in, const int* in_sizes, int n_in,
                              void* d_out, int out_size, void* d_ws, size_t ws_size,
                              hipStream_t stream) {
  (void)in_sizes; (void)n_in; (void)out_size; (void)ws_size;
  const float* x  = (const float*)d_in[0];
  const float* Wf = (const float*)d_in[1];
  const float* Wg = (const float*)d_in[2];
  const float* Wh = (const float*)d_in[3];

  char* ws = (char*)d_ws;
  _Float16* Xt = (_Float16*)(ws);                        // 32 MiB
  _Float16* Ft = (_Float16*)(ws + (size_t)(32  << 20));  //  4 MiB
  _Float16* Gt = (_Float16*)(ws + (size_t)(36  << 20));  //  4 MiB
  _Float16* Hm = (_Float16*)(ws + (size_t)(40  << 20));  // 32 MiB
  _Float16* St = (_Float16*)(ws + (size_t)(72  << 20));  // 64 MiB
  _Float16* Wc = (_Float16*)(ws + (size_t)(136 << 20));  // 640 KiB
  float* out = (float*)d_out;

  k0_transpose<<<BATCH * (CDIM / 32) * (NDIM / 32), 256, 0, stream>>>(x, Xt);
  k0_wconv<<<(ROWS * CDIM) / 256, 256, 0, stream>>>(Wf, Wg, Wh, Wc);
  {
    const int tiles = BATCH * (ROWS / 64) * (NDIM / 16);      // 20480 waves
    k1_proj<<<tiles / 4, 128, 0, stream>>>(Wc, Xt, Ft, Gt, Hm);
  }
  {
    const int tiles = BATCH * (NDIM / 16) * (NDIM / 64);      // 32768 waves
    k2_sim<<<tiles / 4, 128, 0, stream>>>(Ft, Gt, St);
  }
  k3_softmax<<<BATCH * NDIM / 8, 256, 0, stream>>>(St);       // 4096 blocks
  {
    const int tiles = BATCH * (CDIM / 16) * (NDIM / 64);      // 16384 waves
    k4_out<<<tiles / 4, 128, 0, stream>>>(Hm, St, out);
  }
}
